// SimilarClusterEncoder_1116691497734
// MI455X (gfx1250) — matrix operand order repaired and checked by reading, not compiled, not run on hardware
//
#include <hip/hip_runtime.h>
#include <hip/hip_bf16.h>
#include <stdint.h>

typedef __attribute__((ext_vector_type(16))) __bf16 v16bf;
typedef __attribute__((ext_vector_type(8)))  float  v8f;

#define B_    16
#define L_    4096
#define D_    256
#define NC_   128
#define DOUT_ 256

// Packed f32 -> 2xbf16 (RNE) in a single CDNA5 VALU op. The clang builtin is
// not exposed for gfx1250, but the instruction exists; inline asm bypasses
// the frontend gate.
__device__ __forceinline__ unsigned int pack2bf16(float a, float b) {
  unsigned int r;
  asm("v_cvt_pk_bf16_f32 %0, %1, %2" : "=v"(r) : "v"(a), "v"(b));
  return r;
}

union BFrag { v16bf v; unsigned int u[8]; uint4 q[2]; };

// ---------------------------------------------------------------------------
// Prep: P[n][o] = sum_d C[n][d]*W[o][d] + b[o];  csq[n] = ||C[n]||^2
// 128 blocks (one per cluster) x 256 threads (one per output channel).
// ---------------------------------------------------------------------------
__global__ void __launch_bounds__(256)
prep_kernel(const float* __restrict__ C, const float* __restrict__ W,
            const float* __restrict__ bias, float* __restrict__ P,
            float* __restrict__ csq) {
  __shared__ float sC[D_];
  __shared__ float red[256];
  const int n = blockIdx.x, t = threadIdx.x;
  float c = C[n * D_ + t];
  sC[t] = c;
  red[t] = c * c;
  __syncthreads();
  float acc = bias[t];
#pragma unroll 8
  for (int d = 0; d < D_; ++d) acc += sC[d] * W[t * D_ + d];
  P[n * DOUT_ + t] = acc;
  for (int s = 128; s > 0; s >>= 1) {
    if (t < s) red[t] += red[t + s];
    __syncthreads();
  }
  if (t == 0) csq[n] = red[0];
}

// ---------------------------------------------------------------------------
// Main: per 16-token tile compute dots = X_tile @ C^T with bf16 WMMA,
// argmin over (csq[n] - 2*dots[n]), gather P[sel] with mask.
// ---------------------------------------------------------------------------
__global__ void __launch_bounds__(256)
main_kernel(const float* __restrict__ x, const float* __restrict__ C,
            const uint8_t* __restrict__ mask, const float* __restrict__ P,
            const float* __restrict__ csq, float* __restrict__ out) {
  // Centers as packed bf16 pairs, 128 uints per row, XOR-swizzled in 8-uint
  // (32B) chunks so the 512B row stride doesn't collapse onto one bank.
  __shared__ unsigned int ldsC[NC_ * D_ / 2];  // 64 KB

  const int t = threadIdx.x;
  const float2* c2 = (const float2*)C;
#pragma unroll 4
  for (int i = 0; i < 64; ++i) {
    int pairIdx = i * 256 + t;
    float2 f = c2[pairIdx];
    int n  = pairIdx >> 7;
    int k2 = pairIdx & 127;
    int ch = k2 >> 3, w = k2 & 7;
    ldsC[n * 128 + (((ch ^ (n & 15)) << 3) | w)] = pack2bf16(f.x, f.y);
  }
  __syncthreads();

  const int wave = t >> 5;
  const int lane = t & 31;
  const int col  = lane & 15;
  const int hi   = lane >> 4;
  const long m0  = ((long)blockIdx.x * 8 + wave) * 16;  // first token of tile

  float csqv[8];
#pragma unroll
  for (int nt = 0; nt < 8; ++nt) csqv[nt] = csq[nt * 16 + col];

  v8f acc[8];
#pragma unroll
  for (int nt = 0; nt < 8; ++nt) acc[nt] = {};

  const float* xrow = x + (m0 + col) * (long)D_;

#pragma unroll
  for (int kc = 0; kc < 8; ++kc) {
    // A fragment: 16x32 bf16. Lane (l&15)=row M; K runs b..b+15 contiguous,
    // b = kc*32 + 16*hi (matches B's contiguous K order).
    const int ka = kc * 32 + hi * 16;
    const float4* xa = (const float4*)(xrow + ka);
    float4 f0 = xa[0], f1 = xa[1], f2 = xa[2], f3 = xa[3];
    BFrag a;
    a.u[0] = pack2bf16(f0.x, f0.y); a.u[1] = pack2bf16(f0.z, f0.w);
    a.u[2] = pack2bf16(f1.x, f1.y); a.u[3] = pack2bf16(f1.z, f1.w);
    a.u[4] = pack2bf16(f2.x, f2.y); a.u[5] = pack2bf16(f2.z, f2.w);
    a.u[6] = pack2bf16(f3.x, f3.y); a.u[7] = pack2bf16(f3.z, f3.w);

    const int chunk = kc * 2 + hi;  // which 16-K run of this lane
#pragma unroll
    for (int nt = 0; nt < 8; ++nt) {
      // B fragment 32x16: lane holds column n = nt*16+col, contiguous K=16.
      int n = nt * 16 + col;
      const uint4* bp = (const uint4*)&ldsC[n * 128 + ((chunk ^ col) << 3)];
      BFrag b;
      b.q[0] = bp[0];
      b.q[1] = bp[1];
      acc[nt] = __builtin_amdgcn_wmma_f32_16x16x32_bf16(
          false, a.v, false, b.v, (short)0, acc[nt], false, false);
    }
  }

  // Argmin over N=128: acc[nt][r] is element (M = r + 8*hi, N = nt*16 + col).
  // ||x||^2 is row-constant -> argmin over (csq[n] - 2*dot).
  int sel[8];
#pragma unroll
  for (int r = 0; r < 8; ++r) {
    float best = csqv[0] - 2.0f * acc[0][r];
    int   bn   = col;
#pragma unroll
    for (int nt = 1; nt < 8; ++nt) {
      float v = csqv[nt] - 2.0f * acc[nt][r];
      int   n = nt * 16 + col;
      if (v < best) { best = v; bn = n; }
    }
    // butterfly within each 16-lane half (wave32)
#pragma unroll
    for (int m = 1; m < 16; m <<= 1) {
      float ov = __shfl_xor(best, m, 32);
      int   on = __shfl_xor(bn,   m, 32);
      if (ov < best || (ov == best && on < bn)) { best = ov; bn = on; }
    }
    sel[r] = bn;
  }

  // Gather projected centers: token row <- P[sel] * keep (coalesced float4).
  const float4* P4   = (const float4*)P;
  float4*       out4 = (float4*)out;
#pragma unroll
  for (int r = 0; r < 8; ++r) {
    long token = m0 + r + hi * 8;
    float keep = mask[token] ? 0.0f : 1.0f;
    long obase = token * (DOUT_ / 4);
    long pbase = (long)sel[r] * (DOUT_ / 4);
#pragma unroll
    for (int j = 0; j < 4; ++j) {
      int cq = j * 16 + col;
      float4 p = P4[pbase + cq];
      float4 o;
      o.x = p.x * keep; o.y = p.y * keep; o.z = p.z * keep; o.w = p.w * keep;
      out4[obase + cq] = o;
    }
  }
}

extern "C" void kernel_launch(void* const* d_in, const int* in_sizes, int n_in,
                              void* d_out, int out_size, void* d_ws, size_t ws_size,
                              hipStream_t stream) {
  const float*   x    = (const float*)d_in[0];
  const float*   C    = (const float*)d_in[1];
  const float*   W    = (const float*)d_in[2];
  const float*   bias = (const float*)d_in[3];
  const uint8_t* msk  = (const uint8_t*)d_in[4];

  float* P   = (float*)d_ws;                                         // 128*256 f32
  float* csq = (float*)((char*)d_ws + NC_ * DOUT_ * sizeof(float));  // 128 f32

  prep_kernel<<<NC_, 256, 0, stream>>>(C, W, bias, P, csq);

  const int tiles  = (B_ * L_) / 16;   // 4096
  const int blocks = tiles / 8;        // 512 (8 waves/block, 1 tile/wave)
  main_kernel<<<blocks, 256, 0, stream>>>(x, C, msk, P, csq, (float*)d_out);
}
